// CausalSelfAttention_23459111371203
// MI455X (gfx1250) — matrix operand order repaired and checked by reading, not compiled, not run on hardware
//
#include <hip/hip_runtime.h>
#include <math.h>

typedef __attribute__((ext_vector_type(8)))  float  v8f;
typedef __attribute__((ext_vector_type(8)))  __bf16 v8bf;
typedef __attribute__((ext_vector_type(16))) __bf16 v16bf;

#define EMB   1024
#define SEQ   2048
#define NB    2
#define NH    16
#define HD    64
#define MROWS (NB * SEQ)            // 4096

static __device__ __forceinline__ v8f wmma_bf16(v16bf a, v16bf b, v8f c) {
    // D(16x16,f32) = A(16x32,bf16) * B(32x16,bf16) + C
    return __builtin_amdgcn_wmma_f32_16x16x32_bf16(false, a, false, b,
                                                   (short)0, c, false, false);
}

// A-fragment: row-major K-contiguous source; lane needs K-run
// [16*lh, 16*lh+16)  (two 16B loads)
static __device__ __forceinline__ v16bf afrag(const __bf16* p, int lh) {
    v8bf r0 = *(const v8bf*)(p + 16 * lh);
    v8bf r1 = *(const v8bf*)(p + 16 * lh + 8);
    return __builtin_shufflevector(r0, r1, 0, 1, 2, 3, 4, 5, 6, 7,
                                           8, 9, 10, 11, 12, 13, 14, 15);
}

static __device__ __forceinline__ void split2(float v, __bf16& h, __bf16& l) {
    h = (__bf16)v;
    l = (__bf16)(v - (float)h);
}

// ---- CDNA5 async global->LDS copy (ASYNCcnt-tracked, §08_async_tensor) ----
// Generic address of a __shared__ object keeps the LDS byte offset in its
// low 32 bits (LDS aperture decode truncates to addr[31:0]).
static __device__ __forceinline__ void async_ld_b128(const void* gptr, void* lptr) {
    uint32_t lds = (uint32_t)(uintptr_t)lptr;
    asm volatile("global_load_async_to_lds_b128 %0, %1, off"
                 :: "v"(lds), "v"(gptr) : "memory");
}
static __device__ __forceinline__ void wait_async_le2() {
    asm volatile("s_wait_asynccnt 0x2" ::: "memory");
}
static __device__ __forceinline__ void wait_async_le0() {
    asm volatile("s_wait_asynccnt 0x0" ::: "memory");
}

// ---------------------------------------------------------------------------
// Kernel 0: fp32 -> (hi, lo) bf16 split, elementwise
// ---------------------------------------------------------------------------
__global__ __launch_bounds__(256) void split_bf16(const float* __restrict__ src,
                                                  __bf16* __restrict__ hi,
                                                  __bf16* __restrict__ lo, int n) {
    int i = blockIdx.x * 256 + threadIdx.x;
    if (i < n) {
        float v = src[i];
        __bf16 h, l;
        split2(v, h, l);
        hi[i] = h;
        lo[i] = l;
    }
}

// ---------------------------------------------------------------------------
// Kernel 1: qkv = x @ w_qkv^T  via bf16x3 split WMMA.
// Workgroup (8 waves) computes 128(M) x 64(N); the shared 64x32 hi/lo B tile
// is double-buffered in LDS via global_load_async_to_lds_b128 + s_wait_asynccnt.
// Writes Q,K hi/lo bf16 [B,H,S,D] (Q pre-scaled by 1/8, exact) and V
// TRANSPOSED hi/lo bf16 [B,H,D,S].
// ---------------------------------------------------------------------------
__global__ __launch_bounds__(256) void qkv_gemm(
        const __bf16* __restrict__ xh, const __bf16* __restrict__ xl,
        const __bf16* __restrict__ wh, const __bf16* __restrict__ wl,
        __bf16* __restrict__ qhb, __bf16* __restrict__ qlb,
        __bf16* __restrict__ khb, __bf16* __restrict__ klb,
        __bf16* __restrict__ vth, __bf16* __restrict__ vtl) {
    __shared__ __align__(16) __bf16 Bhi[2][64 * 32];
    __shared__ __align__(16) __bf16 Blo[2][64 * 32];

    const int tid  = threadIdx.x;
    const int winb = tid >> 5;
    const int lane = tid & 31;
    const int ml = lane & 15;
    const int lh = lane >> 4;

    const int ntile = blockIdx.x % 48;        // 3E / 64
    const int mblk  = blockIdx.x / 48;        // M / 128
    const int f0 = ntile * 64;
    const int m0 = mblk * 128 + winb * 16;

    // cooperative B loader: thread covers row tid>>2, cols [(tid&3)*8, +8)
    const __bf16* gwh = wh + (size_t)(f0 + (tid >> 2)) * EMB + (tid & 3) * 8;
    const __bf16* gwl = wl + (size_t)(f0 + (tid >> 2)) * EMB + (tid & 3) * 8;
    __bf16* l0h = &Bhi[0][tid * 8];
    __bf16* l1h = &Bhi[1][tid * 8];
    __bf16* l0l = &Blo[0][tid * 8];
    __bf16* l1l = &Blo[1][tid * 8];

    const __bf16* pah = xh + (size_t)(m0 + ml) * EMB;
    const __bf16* pal = xl + (size_t)(m0 + ml) * EMB;

    v8f acc0 = {}, acc1 = {}, acc2 = {}, acc3 = {};

    async_ld_b128(gwh, l0h);
    async_ld_b128(gwl, l0l);
    async_ld_b128(gwh + 32, l1h);
    async_ld_b128(gwl + 32, l1l);

    for (int c = 0; c < EMB / 32; ++c) {
        if (c < EMB / 32 - 1) wait_async_le2(); else wait_async_le0();
        __syncthreads();

        v16bf ah = afrag(pah + 32 * c, lh);
        v16bf al = afrag(pal + 32 * c, lh);
        const __bf16* bh_base = &Bhi[c & 1][0];
        const __bf16* bl_base = &Blo[c & 1][0];
#pragma unroll
        for (int t = 0; t < 4; ++t) {
            v16bf bh = *(const v16bf*)(bh_base + (t * 16 + ml) * 32 + 16 * lh);
            v16bf bl = *(const v16bf*)(bl_base + (t * 16 + ml) * 32 + 16 * lh);
            v8f acc = (t == 0) ? acc0 : (t == 1) ? acc1 : (t == 2) ? acc2 : acc3;
            acc = wmma_bf16(ah, bh, acc);
            acc = wmma_bf16(ah, bl, acc);
            acc = wmma_bf16(al, bh, acc);
            if (t == 0) acc0 = acc; else if (t == 1) acc1 = acc;
            else if (t == 2) acc2 = acc; else acc3 = acc;
        }
        __syncthreads();
        if (c + 2 < EMB / 32) {
            __bf16* dh = ((c & 1) == 0) ? l0h : l1h;
            __bf16* dl = ((c & 1) == 0) ? l0l : l1l;
            async_ld_b128(gwh + 32 * (c + 2), dh);
            async_ld_b128(gwl + 32 * (c + 2), dl);
        }
    }

    const int which = f0 / EMB;               // 0=q, 1=k, 2=v
    const int head  = (f0 % EMB) / HD;
    const int b     = m0 / SEQ;
    const int s0    = m0 % SEQ;
    const float sc  = (which == 0) ? 0.125f : 1.0f;   // fold 1/sqrt(64) into Q
    __bf16* dsth = (which == 0) ? qhb : ((which == 1) ? khb : vth);
    __bf16* dstl = (which == 0) ? qlb : ((which == 1) ? klb : vtl);
    const size_t bh_ = (size_t)b * NH + head;

#pragma unroll
    for (int t = 0; t < 4; ++t) {
        v8f a = (t == 0) ? acc0 : (t == 1) ? acc1 : (t == 2) ? acc2 : acc3;
        const int d = t * 16 + ml;
#pragma unroll
        for (int i = 0; i < 8; ++i) {
            const int s = s0 + i + 8 * lh;
            __bf16 h, l;
            split2(a[i] * sc, h, l);
            const size_t addr = (which == 2)
                              ? (bh_ * HD + d) * SEQ + s     // V: [B,H,D,S]
                              : (bh_ * SEQ + s) * HD + d;    // Q,K: [B,H,S,D]
            dsth[addr] = h;
            dstl[addr] = l;
        }
    }
}

// ---------------------------------------------------------------------------
// Kernel 2: causal flash attention, bf16x3 WMMA, 32-key blocks.
// One wave owns 16 query rows. P re-laid C->A via per-wave LDS tile
// (DS ops are in-order within a wave). Output: attn hi/lo bf16 [B,S,E].
// ---------------------------------------------------------------------------
__global__ __launch_bounds__(256) void flash_attn(
        const __bf16* __restrict__ qhb, const __bf16* __restrict__ qlb,
        const __bf16* __restrict__ khb, const __bf16* __restrict__ klb,
        const __bf16* __restrict__ vth, const __bf16* __restrict__ vtl,
        __bf16* __restrict__ ahb, __bf16* __restrict__ alb) {
    __shared__ __align__(16) __bf16 PhiS[8][16 * 32];
    __shared__ __align__(16) __bf16 PloS[8][16 * 32];

    const int wave = (blockIdx.x * blockDim.x + threadIdx.x) >> 5;
    const int winb = threadIdx.x >> 5;
    const int lane = threadIdx.x & 31;
    const int ml = lane & 15;
    const int lh = lane >> 4;

    const int qt = wave & 127;                 // S/16 query tiles
    const int hh = (wave >> 7) & 15;
    const int b  = wave >> 11;

    const size_t base = ((size_t)b * NH + hh) * SEQ * HD;  // same extent both layouts
    const __bf16* Qh = qhb + base;
    const __bf16* Ql = qlb + base;
    const __bf16* Kh = khb + base;
    const __bf16* Kl = klb + base;
    const __bf16* Vh = vth + base;             // [D, S] within head
    const __bf16* Vl = vtl + base;

    v16bf qfh[2], qfl[2];
    {
        const __bf16* ph = Qh + (size_t)(qt * 16 + ml) * HD;
        const __bf16* pl = Ql + (size_t)(qt * 16 + ml) * HD;
#pragma unroll
        for (int c = 0; c < 2; ++c) {
            qfh[c] = afrag(ph + 32 * c, lh);
            qfl[c] = afrag(pl + 32 * c, lh);
        }
    }

    float mrow[8], lrow[8];
#pragma unroll
    for (int i = 0; i < 8; ++i) { mrow[i] = -INFINITY; lrow[i] = 0.0f; }
    v8f o0 = {}, o1 = {}, o2 = {}, o3 = {};
    __bf16* Ph = PhiS[winb];
    __bf16* Pl = PloS[winb];

    const int kbmax = qt >> 1;                 // 32-key blocks under causal limit
    for (int kb = 0; kb <= kbmax; ++kb) {
        v8f s0 = {}, s1 = {};
        const __bf16* k0h = Kh + (size_t)(kb * 32 + ml) * HD + 16 * lh;
        const __bf16* k0l = Kl + (size_t)(kb * 32 + ml) * HD + 16 * lh;
#pragma unroll
        for (int c = 0; c < 2; ++c) {
            v16bf bh0 = *(const v16bf*)(k0h + 32 * c);
            v16bf bl0 = *(const v16bf*)(k0l + 32 * c);
            v16bf bh1 = *(const v16bf*)(k0h + 16 * HD + 32 * c);
            v16bf bl1 = *(const v16bf*)(k0l + 16 * HD + 32 * c);
            s0 = wmma_bf16(qfh[c], bh0, s0);
            s0 = wmma_bf16(qfh[c], bl0, s0);
            s0 = wmma_bf16(qfl[c], bh0, s0);
            s1 = wmma_bf16(qfh[c], bh1, s1);
            s1 = wmma_bf16(qfh[c], bl1, s1);
            s1 = wmma_bf16(qfl[c], bh1, s1);
        }

        if (kb == kbmax) {
#pragma unroll
            for (int i = 0; i < 8; ++i) {
                const int q  = qt * 16 + i + 8 * lh;
                const int k0 = kb * 32 + ml;
                if (k0 > q)      s0[i] = -INFINITY;
                if (k0 + 16 > q) s1[i] = -INFINITY;
            }
        }

        float alpha[8];
#pragma unroll
        for (int i = 0; i < 8; ++i) {
            float v = fmaxf(s0[i], s1[i]);
            v = fmaxf(v, __shfl_xor(v, 1, 32));
            v = fmaxf(v, __shfl_xor(v, 2, 32));
            v = fmaxf(v, __shfl_xor(v, 4, 32));
            v = fmaxf(v, __shfl_xor(v, 8, 32));
            float mnew = fmaxf(mrow[i], v);
            alpha[i] = __expf(mrow[i] - mnew);
            mrow[i]  = mnew;
            float p0 = __expf(s0[i] - mnew);
            float p1 = __expf(s1[i] - mnew);
            s0[i] = p0;
            s1[i] = p1;
            float rs = p0 + p1;
            rs += __shfl_xor(rs, 1, 32);
            rs += __shfl_xor(rs, 2, 32);
            rs += __shfl_xor(rs, 4, 32);
            rs += __shfl_xor(rs, 8, 32);
            lrow[i] = lrow[i] * alpha[i] + rs;
        }
#pragma unroll
        for (int i = 0; i < 8; ++i) {
            o0[i] *= alpha[i]; o1[i] *= alpha[i];
            o2[i] *= alpha[i]; o3[i] *= alpha[i];
        }

#pragma unroll
        for (int i = 0; i < 8; ++i) {
            const int r = (i + 8 * lh) * 32;
            __bf16 h, l;
            split2(s0[i], h, l);
            Ph[r + ml] = h;      Pl[r + ml] = l;
            split2(s1[i], h, l);
            Ph[r + 16 + ml] = h; Pl[r + 16 + ml] = l;
        }

        v16bf pah = afrag(Ph + ml * 32, lh);
        v16bf pal = afrag(Pl + ml * 32, lh);
        const __bf16* vh0 = Vh + (size_t)ml * SEQ + kb * 32 + 16 * lh;
        const __bf16* vl0 = Vl + (size_t)ml * SEQ + kb * 32 + 16 * lh;
        {
            v16bf bh = *(const v16bf*)(vh0);
            v16bf bl = *(const v16bf*)(vl0);
            o0 = wmma_bf16(pah, bh, o0);
            o0 = wmma_bf16(pah, bl, o0);
            o0 = wmma_bf16(pal, bh, o0);
        }
        {
            v16bf bh = *(const v16bf*)(vh0 + (size_t)16 * SEQ);
            v16bf bl = *(const v16bf*)(vl0 + (size_t)16 * SEQ);
            o1 = wmma_bf16(pah, bh, o1);
            o1 = wmma_bf16(pah, bl, o1);
            o1 = wmma_bf16(pal, bh, o1);
        }
        {
            v16bf bh = *(const v16bf*)(vh0 + (size_t)32 * SEQ);
            v16bf bl = *(const v16bf*)(vl0 + (size_t)32 * SEQ);
            o2 = wmma_bf16(pah, bh, o2);
            o2 = wmma_bf16(pah, bl, o2);
            o2 = wmma_bf16(pal, bh, o2);
        }
        {
            v16bf bh = *(const v16bf*)(vh0 + (size_t)48 * SEQ);
            v16bf bl = *(const v16bf*)(vl0 + (size_t)48 * SEQ);
            o3 = wmma_bf16(pah, bh, o3);
            o3 = wmma_bf16(pah, bl, o3);
            o3 = wmma_bf16(pal, bh, o3);
        }
    }

    const size_t colbase = (size_t)hh * HD;
#pragma unroll
    for (int i = 0; i < 8; ++i) {
        const float inv = 1.0f / lrow[i];
        const size_t row = ((size_t)b * SEQ + qt * 16 + i + 8 * lh) * EMB + colbase;
        __bf16 h, l;
        split2(o0[i] * inv, h, l); ahb[row +  0 + ml] = h; alb[row +  0 + ml] = l;
        split2(o1[i] * inv, h, l); ahb[row + 16 + ml] = h; alb[row + 16 + ml] = l;
        split2(o2[i] * inv, h, l); ahb[row + 32 + ml] = h; alb[row + 32 + ml] = l;
        split2(o3[i] * inv, h, l); ahb[row + 48 + ml] = h; alb[row + 48 + ml] = l;
    }
}

// ---------------------------------------------------------------------------
// Kernel 3: out = attn @ w_out^T + b_out  (bf16x3 WMMA, async-LDS B staging)
// ---------------------------------------------------------------------------
__global__ __launch_bounds__(256) void out_gemm(
        const __bf16* __restrict__ ah, const __bf16* __restrict__ al,
        const __bf16* __restrict__ wh, const __bf16* __restrict__ wl,
        const float* __restrict__ bias, float* __restrict__ out) {
    __shared__ __align__(16) __bf16 Bhi[2][64 * 32];
    __shared__ __align__(16) __bf16 Blo[2][64 * 32];

    const int tid  = threadIdx.x;
    const int winb = tid >> 5;
    const int lane = tid & 31;
    const int ml = lane & 15;
    const int lh = lane >> 4;

    const int ntile = blockIdx.x & 15;        // E / 64
    const int mblk  = blockIdx.x >> 4;        // M / 128
    const int f0 = ntile * 64;
    const int m0 = mblk * 128 + winb * 16;

    const __bf16* gwh = wh + (size_t)(f0 + (tid >> 2)) * EMB + (tid & 3) * 8;
    const __bf16* gwl = wl + (size_t)(f0 + (tid >> 2)) * EMB + (tid & 3) * 8;
    __bf16* l0h = &Bhi[0][tid * 8];
    __bf16* l1h = &Bhi[1][tid * 8];
    __bf16* l0l = &Blo[0][tid * 8];
    __bf16* l1l = &Blo[1][tid * 8];

    const __bf16* pah = ah + (size_t)(m0 + ml) * EMB;
    const __bf16* pal = al + (size_t)(m0 + ml) * EMB;

    v8f acc0 = {}, acc1 = {}, acc2 = {}, acc3 = {};

    async_ld_b128(gwh, l0h);
    async_ld_b128(gwl, l0l);
    async_ld_b128(gwh + 32, l1h);
    async_ld_b128(gwl + 32, l1l);

    for (int c = 0; c < EMB / 32; ++c) {
        if (c < EMB / 32 - 1) wait_async_le2(); else wait_async_le0();
        __syncthreads();

        v16bf a_h = afrag(pah + 32 * c, lh);
        v16bf a_l = afrag(pal + 32 * c, lh);
        const __bf16* bh_base = &Bhi[c & 1][0];
        const __bf16* bl_base = &Blo[c & 1][0];
#pragma unroll
        for (int t = 0; t < 4; ++t) {
            v16bf bh = *(const v16bf*)(bh_base + (t * 16 + ml) * 32 + 16 * lh);
            v16bf bl = *(const v16bf*)(bl_base + (t * 16 + ml) * 32 + 16 * lh);
            v8f acc = (t == 0) ? acc0 : (t == 1) ? acc1 : (t == 2) ? acc2 : acc3;
            acc = wmma_bf16(a_h, bh, acc);
            acc = wmma_bf16(a_h, bl, acc);
            acc = wmma_bf16(a_l, bh, acc);
            if (t == 0) acc0 = acc; else if (t == 1) acc1 = acc;
            else if (t == 2) acc2 = acc; else acc3 = acc;
        }
        __syncthreads();
        if (c + 2 < EMB / 32) {
            __bf16* dh = ((c & 1) == 0) ? l0h : l1h;
            __bf16* dl = ((c & 1) == 0) ? l0l : l1l;
            async_ld_b128(gwh + 32 * (c + 2), dh);
            async_ld_b128(gwl + 32 * (c + 2), dl);
        }
    }

    const float b0 = bias[f0 +  0 + ml];
    const float b1 = bias[f0 + 16 + ml];
    const float b2 = bias[f0 + 32 + ml];
    const float b3 = bias[f0 + 48 + ml];
#pragma unroll
    for (int i = 0; i < 8; ++i) {
        float* row = out + (size_t)(m0 + i + 8 * lh) * EMB + f0;
        row[ 0 + ml] = acc0[i] + b0;
        row[16 + ml] = acc1[i] + b1;
        row[32 + ml] = acc2[i] + b2;
        row[48 + ml] = acc3[i] + b3;
    }
}

// ---------------------------------------------------------------------------
extern "C" void kernel_launch(void* const* d_in, const int* in_sizes, int n_in,
                              void* d_out, int out_size, void* d_ws, size_t ws_size,
                              hipStream_t stream) {
    const float* x     = (const float*)d_in[0];   // [2,2048,1024]
    const float* w_qkv = (const float*)d_in[1];   // [3072,1024]
    const float* w_out = (const float*)d_in[2];   // [1024,1024]
    const float* b_out = (const float*)d_in[3];   // [1024]
    float* out = (float*)d_out;                   // [2,2048,1024]

    const size_t nx    = (size_t)MROWS * EMB;     // 4,194,304
    const size_t nwqkv = (size_t)3 * EMB * EMB;   // 3,145,728
    const size_t nwout = (size_t)EMB * EMB;       // 1,048,576

    __bf16* p   = (__bf16*)d_ws;
    __bf16* xh  = p;            __bf16* xl  = xh  + nx;
    __bf16* wqh = xl + nx;      __bf16* wql = wqh + nwqkv;
    __bf16* woh = wql + nwqkv;  __bf16* wol = woh + nwout;
    __bf16* qh  = wol + nwout;  __bf16* ql  = qh  + nx;
    __bf16* kh  = ql + nx;      __bf16* kl  = kh  + nx;
    __bf16* vth = kl + nx;      __bf16* vtl = vth + nx;
    // attn hi/lo alias the x split (x is dead after qkv_gemm)
    __bf16* ath = xh;
    __bf16* atl = xl;

    split_bf16<<<(int)(nx    / 256), 256, 0, stream>>>(x,     xh,  xl,  (int)nx);
    split_bf16<<<(int)(nwqkv / 256), 256, 0, stream>>>(w_qkv, wqh, wql, (int)nwqkv);
    split_bf16<<<(int)(nwout / 256), 256, 0, stream>>>(w_out, woh, wol, (int)nwout);

    // 32 M-blocks * 48 N-tiles = 1536 workgroups of 8 waves (128x64 each)
    qkv_gemm<<<1536, 256, 0, stream>>>(xh, xl, wqh, wql, qh, ql, kh, kl, vth, vtl);
    // 2*16*128 = 4096 waves -> 512 blocks
    flash_attn<<<512, 256, 0, stream>>>(qh, ql, kh, kl, vth, vtl, ath, atl);
    // 32 M-blocks * 16 N-tiles = 512 workgroups
    out_gemm<<<512, 256, 0, stream>>>(ath, atl, woh, wol, b_out, out);
}
